// CrossAttention_38336878084674
// MI455X (gfx1250) — compile-verified, matrix operand order checked by reading, not run on hardware
//
#include <hip/hip_runtime.h>

#define BB 2
#define NQ 2048
#define MK 2048
#define QDIM 1024
#define CDIM 1024
#define HEADS 16
#define DHEAD 64
#define INNER 1024
#define ATTN_SCALE 0.125f   // 64^-0.5
#define NEG_BIG (-1.0e30f)

typedef __attribute__((ext_vector_type(16))) __bf16 v16bf;
typedef __attribute__((ext_vector_type(8)))  float  v8f;
typedef int v4i __attribute__((vector_size(16)));   // matches builtin param type

union FragB16 { v16bf v; unsigned short u[16]; };

// ---- async global->LDS path (gfx1250), guarded so compile never breaks -----
#if defined(__HIP_DEVICE_COMPILE__) &&                                        \
    __has_builtin(__builtin_amdgcn_global_load_async_to_lds_b128) &&          \
    __has_builtin(__builtin_amdgcn_s_wait_asynccnt)
#define HAVE_ASYNC_LDS 1
#else
#define HAVE_ASYNC_LDS 0
#endif

#if HAVE_ASYNC_LDS
typedef __attribute__((address_space(1))) v4i* gptr_v4i;
typedef __attribute__((address_space(3))) v4i* lptr_v4i;
#endif

// LDS tile strides (ushorts). Rows are 16B-multiples (async B128 alignment)
// yet odd multiples of 4 banks, keeping fragment reads low-conflict.
#define A_STRIDE 40   // 64 rows x 32 k  (row = 80B)
#define B_STRIDE 72   // 32 k   x 64 col (row = 144B)

__device__ __forceinline__ void async_cp16(const unsigned short* g,
                                           unsigned short* l) {
#if HAVE_ASYNC_LDS
  auto gp = (__attribute__((address_space(1))) void*)(void*)g;
  auto lp = (__attribute__((address_space(3))) void*)(void*)l;
  __builtin_amdgcn_global_load_async_to_lds_b128((gptr_v4i)gp, (lptr_v4i)lp, 0, 0);
#else
  (void)g; (void)l;
#endif
}

__device__ __forceinline__ unsigned short f32_to_bf16(float f) {
  unsigned int x = __float_as_uint(f);
  unsigned int r = 0x7FFFu + ((x >> 16) & 1u);   // round-to-nearest-even
  return (unsigned short)((x + r) >> 16);
}

__device__ __forceinline__ v8f v8f_zero() {
  v8f z;
#pragma unroll
  for (int i = 0; i < 8; ++i) z[i] = 0.0f;
  return z;
}

__device__ __forceinline__ v8f wmma_bf16(const v16bf& a, const v16bf& b, v8f c) {
  // D = A(16x32) * B(32x16) + C, f32 accumulate
  return __builtin_amdgcn_wmma_f32_16x16x32_bf16(false, a, false, b, (short)0, c,
                                                 false, false);
}

// ---------------------------------------------------------------- fp32 -> bf16
__global__ __launch_bounds__(256) void cvt(const float* __restrict__ in,
                                           unsigned short* __restrict__ out, int n) {
  int i = blockIdx.x * 256 + threadIdx.x;
  if (i < n) out[i] = f32_to_bf16(in[i]);
}

// -------------------------------------------------- V[b,m,h*64+d] -> Vt[b,h,d,m]
__global__ __launch_bounds__(256) void transpose_v(const unsigned short* __restrict__ V,
                                                   unsigned short* __restrict__ Vt, int n) {
  int i = blockIdx.x * 256 + threadIdx.x;
  if (i >= n) return;
  int m = i & (MK - 1);
  int r = i / MK;
  int d = r % DHEAD; r /= DHEAD;
  int h = r % HEADS;
  int b = r / HEADS;
  Vt[i] = V[((size_t)b * MK + m) * INNER + h * DHEAD + d];
}

// ------------------------------------------------------------------- bf16 GEMM
// C[Mrows x Nc] = A[Mrows x Kc] * B[Kc x Nc]; 64x64 block tile, 32x32 wave tile.
// Async double-buffered LDS pipeline on gfx1250; sync single-buffer fallback.
template <bool F32OUT>
__global__ __launch_bounds__(128)
void gemm_bf16(const unsigned short* __restrict__ A,
               const unsigned short* __restrict__ Bm,
               const float* __restrict__ bias, float scale,
               unsigned short* __restrict__ Cb, float* __restrict__ Cf,
               int Mrows, int Nc, int Kc) {
  const int tid  = threadIdx.x;
  const int lane = tid & 31;
  const int w    = tid >> 5;
  const int ln   = lane & 15;
  const int lh   = lane >> 4;
  const int m0   = blockIdx.y * 64;
  const int n0   = blockIdx.x * 64;
  const int wm   = (w >> 1) * 32;
  const int wn   = (w & 1) * 32;

  v8f acc[2][2];
#pragma unroll
  for (int i = 0; i < 2; ++i)
#pragma unroll
    for (int j = 0; j < 2; ++j) acc[i][j] = v8f_zero();

#if HAVE_ASYNC_LDS
  // ---------------- async double-buffered pipeline ----------------
  __shared__ unsigned short As[2][64][A_STRIDE];
  __shared__ unsigned short Bs[2][32][B_STRIDE];

  // Each thread issues 2 x 16B chunks of A and 2 x 16B chunks of B per tile.
  auto stage = [&](int buf, int k0) {
#pragma unroll
    for (int i = 0; i < 2; ++i) {
      int c   = i * 128 + tid;          // A: 256 chunks, row = 64B = 4 chunks
      int row = c >> 2, ch = c & 3;
      async_cp16(&A[(size_t)(m0 + row) * Kc + k0 + ch * 8],
                 &As[buf][row][ch * 8]);
    }
#pragma unroll
    for (int i = 0; i < 2; ++i) {
      int c   = i * 128 + tid;          // B: 256 chunks, row = 128B = 8 chunks
      int row = c >> 3, ch = c & 7;
      async_cp16(&Bm[(size_t)(k0 + row) * Nc + n0 + ch * 8],
                 &Bs[buf][row][ch * 8]);
    }
  };

  stage(0, 0);
  for (int k0 = 0, it = 0; k0 < Kc; k0 += 32, ++it) {
    const int cur = it & 1;
    __syncthreads();                      // buffer cur^1 readers are done
    if (k0 + 32 < Kc) {
      stage(cur ^ 1, k0 + 32);            // prefetch next tile
      __builtin_amdgcn_s_wait_asynccnt(4);  // older group (cur) has landed
    } else {
      __builtin_amdgcn_s_wait_asynccnt(0);
    }
    __syncthreads();                      // everyone's cur data visible
    unsigned short (*Asc)[A_STRIDE] = As[cur];
    unsigned short (*Bsc)[B_STRIDE] = Bs[cur];
#else
  // ---------------- synchronous fallback ----------------
  __shared__ unsigned short As1[64][A_STRIDE];
  __shared__ unsigned short Bs1[32][B_STRIDE];
  for (int k0 = 0; k0 < Kc; k0 += 32) {
    __syncthreads();
#pragma unroll
    for (int i = 0; i < 8; ++i) {
      int d   = i * 128 + tid;
      int row = d >> 4, c2 = d & 15;
      unsigned int val =
          *(const unsigned int*)&A[(size_t)(m0 + row) * Kc + k0 + c2 * 2];
      As1[row][c2 * 2]     = (unsigned short)(val & 0xFFFFu);
      As1[row][c2 * 2 + 1] = (unsigned short)(val >> 16);
    }
#pragma unroll
    for (int i = 0; i < 8; ++i) {
      int d   = i * 128 + tid;
      int row = d >> 5, c2 = d & 31;
      unsigned int val =
          *(const unsigned int*)&Bm[(size_t)(k0 + row) * Nc + n0 + c2 * 2];
      Bs1[row][c2 * 2]     = (unsigned short)(val & 0xFFFFu);
      Bs1[row][c2 * 2 + 1] = (unsigned short)(val >> 16);
    }
    if (k0 + 32 < Kc) {
      __builtin_prefetch(&A[(size_t)(m0 + (tid >> 1)) * Kc + k0 + 32], 0, 0);
      __builtin_prefetch(&Bm[(size_t)(k0 + 32 + (tid >> 2)) * Nc + n0], 0, 0);
    }
    __syncthreads();
    unsigned short (*Asc)[A_STRIDE] = As1;
    unsigned short (*Bsc)[B_STRIDE] = Bs1;
#endif

    FragB16 af[2], bfr[2];
#pragma unroll
    for (int i = 0; i < 2; ++i) {
      int row = wm + 16 * i + ln;
#pragma unroll
      for (int j = 0; j < 8; ++j) {
        int kk = (j >> 2) * 16 + lh * 8 + (j & 3) * 2;
        af[i].u[2 * j]     = Asc[row][kk];
        af[i].u[2 * j + 1] = Asc[row][kk + 1];
      }
    }
#pragma unroll
    for (int jn = 0; jn < 2; ++jn) {
      int col = wn + 16 * jn + ln;
#pragma unroll
      for (int j = 0; j < 8; ++j) {
        int kk = (j >> 2) * 16 + lh * 8 + (j & 3) * 2;
        bfr[jn].u[2 * j]     = Bsc[kk][col];
        bfr[jn].u[2 * j + 1] = Bsc[kk + 1][col];
      }
    }
#pragma unroll
    for (int i = 0; i < 2; ++i)
#pragma unroll
      for (int j = 0; j < 2; ++j)
        acc[i][j] = wmma_bf16(af[i].v, bfr[j].v, acc[i][j]);
  }

  // epilogue: C layout -> vgpr r = row r + 8*lh, column = ln
#pragma unroll
  for (int i = 0; i < 2; ++i)
#pragma unroll
    for (int j = 0; j < 2; ++j)
#pragma unroll
      for (int r = 0; r < 8; ++r) {
        int row = m0 + wm + 16 * i + r + 8 * lh;
        int col = n0 + wn + 16 * j + ln;
        float v = acc[i][j][r] * scale;
        if (F32OUT)
          Cf[(size_t)row * Nc + col] = v + (bias ? bias[col] : 0.0f);
        else
          Cb[(size_t)row * Nc + col] = f32_to_bf16(v);
      }
}

// ------------------------------------------------------------- flash attention
// grid = (NQ/64, BB*HEADS), block = 128 (4 waves, 16 query rows per wave)
__global__ __launch_bounds__(128)
void flash_attn(const unsigned short* __restrict__ Q,
                const unsigned short* __restrict__ Kmat,
                const unsigned short* __restrict__ Vt,
                const unsigned char* __restrict__ mask,
                unsigned short* __restrict__ O) {
  __shared__ unsigned short pls[4][16][40];   // per-wave P relayout buffer

  const int bh = blockIdx.y;
  const int b = bh / HEADS, h = bh % HEADS;
  const int tid  = threadIdx.x;
  const int w    = tid >> 5;
  const int lane = tid & 31;
  const int ln   = lane & 15;
  const int lh   = lane >> 4;
  const int qrow = blockIdx.x * 64 + w * 16 + ln;

  // Q A-fragments (16 rows x 64 dims -> two 16x32 fragments), pre-scaled
  FragB16 qa[2];
  {
    const size_t qbase = ((size_t)b * NQ + qrow) * INNER + h * DHEAD;
#pragma unroll
    for (int s = 0; s < 2; ++s)
#pragma unroll
      for (int j = 0; j < 8; ++j) {
        int kk = (j >> 2) * 16 + lh * 8 + (j & 3) * 2;
        unsigned int val = *(const unsigned int*)&Q[qbase + s * 32 + kk];
        qa[s].u[2 * j]     = (unsigned short)(val & 0xFFFFu);
        qa[s].u[2 * j + 1] = (unsigned short)(val >> 16);
      }
  }

  float mrow[8], lrow[8];
  v8f o[4];
#pragma unroll
  for (int r = 0; r < 8; ++r) { mrow[r] = NEG_BIG; lrow[r] = 0.0f; }
#pragma unroll
  for (int t = 0; t < 4; ++t) o[t] = v8f_zero();

  for (int kb = 0; kb < MK; kb += 32) {
    // mask bytes first (tiny, independent)
    const bool mk0 = mask[(size_t)b * MK + kb + ln] != 0;
    const bool mk1 = mask[(size_t)b * MK + kb + 16 + ln] != 0;

    // K^T B-fragments: kt[key subtile][dim chunk]
    FragB16 kt[2][2];
#pragma unroll
    for (int t = 0; t < 2; ++t) {
      const size_t kbase = ((size_t)b * MK + kb + t * 16 + ln) * INNER + h * DHEAD;
#pragma unroll
      for (int s = 0; s < 2; ++s)
#pragma unroll
        for (int j = 0; j < 8; ++j) {
          int kk = (j >> 2) * 16 + lh * 8 + (j & 3) * 2;
          unsigned int val = *(const unsigned int*)&Kmat[kbase + s * 32 + kk];
          kt[t][s].u[2 * j]     = (unsigned short)(val & 0xFFFFu);
          kt[t][s].u[2 * j + 1] = (unsigned short)(val >> 16);
        }
    }
    v8f s0 = wmma_bf16(qa[0].v, kt[0][0].v, wmma_bf16(qa[1].v, kt[0][1].v, v8f_zero()));
    v8f s1 = wmma_bf16(qa[0].v, kt[1][0].v, wmma_bf16(qa[1].v, kt[1][1].v, v8f_zero()));

    // hoist V-fragment loads: independent of softmax -> overlaps VALU below
    FragB16 vbf[4];
#pragma unroll
    for (int t = 0; t < 4; ++t) {
      const size_t vbase =
          (((size_t)b * HEADS + h) * DHEAD + t * 16 + ln) * MK + kb;
#pragma unroll
      for (int j = 0; j < 8; ++j) {
        int kk = (j >> 2) * 16 + lh * 8 + (j & 3) * 2;
        unsigned int val = *(const unsigned int*)&Vt[vbase + kk];
        vbf[t].u[2 * j]     = (unsigned short)(val & 0xFFFFu);
        vbf[t].u[2 * j + 1] = (unsigned short)(val >> 16);
      }
    }

#pragma unroll
    for (int r = 0; r < 8; ++r) {
      if (!mk0) s0[r] = NEG_BIG;
      if (!mk1) s1[r] = NEG_BIG;
    }

    // online softmax: row stats replicated across each 16-lane half
    float cm[8];
#pragma unroll
    for (int r = 0; r < 8; ++r) cm[r] = fmaxf(s0[r], s1[r]);
#pragma unroll
    for (int m = 1; m < 16; m <<= 1)
#pragma unroll
      for (int r = 0; r < 8; ++r) cm[r] = fmaxf(cm[r], __shfl_xor(cm[r], m, 16));

    float alpha[8], p0[8], p1[8], cs[8];
#pragma unroll
    for (int r = 0; r < 8; ++r) {
      float mn = fmaxf(mrow[r], cm[r]);
      alpha[r] = __expf(mrow[r] - mn);
      mrow[r]  = mn;
      p0[r] = __expf(s0[r] - mn);
      p1[r] = __expf(s1[r] - mn);
      cs[r] = p0[r] + p1[r];
    }
#pragma unroll
    for (int m = 1; m < 16; m <<= 1)
#pragma unroll
      for (int r = 0; r < 8; ++r) cs[r] += __shfl_xor(cs[r], m, 16);
#pragma unroll
    for (int r = 0; r < 8; ++r) lrow[r] = lrow[r] * alpha[r] + cs[r];
#pragma unroll
    for (int t = 0; t < 4; ++t)
#pragma unroll
      for (int r = 0; r < 8; ++r) o[t][r] *= alpha[r];

    // re-layout P: C-format (cols on lanes) -> A-format (rows on lanes) via LDS
#pragma unroll
    for (int r = 0; r < 8; ++r) {
      int row = r + 8 * lh;
      pls[w][row][ln]      = f32_to_bf16(p0[r]);
      pls[w][row][16 + ln] = f32_to_bf16(p1[r]);
    }
    FragB16 pa;
#pragma unroll
    for (int j = 0; j < 8; ++j) {
      int kk = (j >> 2) * 16 + lh * 8 + (j & 3) * 2;
      pa.u[2 * j]     = pls[w][ln][kk];
      pa.u[2 * j + 1] = pls[w][ln][kk + 1];
    }

    // O += P(16x32) * V(32x16) per 16-dim tile
#pragma unroll
    for (int t = 0; t < 4; ++t) o[t] = wmma_bf16(pa.v, vbf[t].v, o[t]);
  }

  // normalize + store bf16 O[b, q, h*64+d]
  float inv[8];
#pragma unroll
  for (int r = 0; r < 8; ++r) inv[r] = (lrow[r] > 0.0f) ? 1.0f / lrow[r] : 0.0f;
#pragma unroll
  for (int t = 0; t < 4; ++t)
#pragma unroll
    for (int r = 0; r < 8; ++r) {
      int row = blockIdx.x * 64 + w * 16 + r + 8 * lh;
      O[((size_t)b * NQ + row) * INNER + h * DHEAD + t * 16 + ln] =
          f32_to_bf16(o[t][r] * inv[r]);
    }
}

// ------------------------------------------------------------------- launcher
extern "C" void kernel_launch(void* const* d_in, const int* in_sizes, int n_in,
                              void* d_out, int out_size, void* d_ws, size_t ws_size,
                              hipStream_t stream) {
  const float*         x    = (const float*)d_in[0];
  const float*         ctx  = (const float*)d_in[1];
  const unsigned char* mask = (const unsigned char*)d_in[2];
  const float*         Wq   = (const float*)d_in[3];
  const float*         Wk   = (const float*)d_in[4];
  const float*         Wv   = (const float*)d_in[5];
  const float*         Wo   = (const float*)d_in[6];
  const float*         bo   = (const float*)d_in[7];
  float*               out  = (float*)d_out;

  char* ws = (char*)d_ws;
  const size_t MB = 1024 * 1024;
  unsigned short* xb  = (unsigned short*)(ws + 0 * MB);    // 8 MB
  unsigned short* cb  = (unsigned short*)(ws + 8 * MB);    // 8 MB
  unsigned short* wqb = (unsigned short*)(ws + 16 * MB);   // 2 MB
  unsigned short* wkb = (unsigned short*)(ws + 18 * MB);   // 2 MB
  unsigned short* wvb = (unsigned short*)(ws + 20 * MB);   // 2 MB
  unsigned short* wob = (unsigned short*)(ws + 22 * MB);   // 2 MB
  unsigned short* Qb  = (unsigned short*)(ws + 24 * MB);   // 8 MB
  unsigned short* Kb  = (unsigned short*)(ws + 32 * MB);   // 8 MB
  unsigned short* Vb  = (unsigned short*)(ws + 40 * MB);   // 8 MB
  unsigned short* Vtb = (unsigned short*)(ws + 48 * MB);   // 8 MB
  unsigned short* Ob  = (unsigned short*)(ws + 56 * MB);   // 8 MB (64 MB total)

  const int nx = BB * NQ * QDIM;      // 4,194,304
  const int nw = QDIM * INNER;        // 1,048,576
  cvt<<<(nx + 255) / 256, 256, 0, stream>>>(x,   xb,  nx);
  cvt<<<(nx + 255) / 256, 256, 0, stream>>>(ctx, cb,  nx);
  cvt<<<(nw + 255) / 256, 256, 0, stream>>>(Wq,  wqb, nw);
  cvt<<<(nw + 255) / 256, 256, 0, stream>>>(Wk,  wkb, nw);
  cvt<<<(nw + 255) / 256, 256, 0, stream>>>(Wv,  wvb, nw);
  cvt<<<(nw + 255) / 256, 256, 0, stream>>>(Wo,  wob, nw);

  dim3 gg(INNER / 64, (BB * NQ) / 64);
  gemm_bf16<false><<<gg, 128, 0, stream>>>(xb, wqb, nullptr, ATTN_SCALE, Qb, nullptr,
                                           BB * NQ, INNER, QDIM);
  gemm_bf16<false><<<gg, 128, 0, stream>>>(cb, wkb, nullptr, 1.0f, Kb, nullptr,
                                           BB * MK, INNER, CDIM);
  gemm_bf16<false><<<gg, 128, 0, stream>>>(cb, wvb, nullptr, 1.0f, Vb, nullptr,
                                           BB * MK, INNER, CDIM);

  const int nv = BB * HEADS * DHEAD * MK;
  transpose_v<<<(nv + 255) / 256, 256, 0, stream>>>(Vb, Vtb, nv);

  dim3 ga(NQ / 64, BB * HEADS);
  flash_attn<<<ga, 128, 0, stream>>>(Qb, Kb, Vtb, mask, Ob);

  dim3 go(QDIM / 64, (BB * NQ) / 64);
  gemm_bf16<true><<<go, 128, 0, stream>>>(Ob, wob, bo, 1.0f, nullptr, out,
                                          BB * NQ, QDIM, INNER);
}
